// EnhancedCGCNN_23192823398614
// MI455X (gfx1250) — matrix operand order repaired and checked by reading, not compiled, not run on hardware
//
#include <hip/hip_runtime.h>
#include <hip/hip_bf16.h>

#define N_NODES 20000
#define N_EDGES 400000
#define N_GRAPHS 16
#define HDIM 256
#define NLAYERS 6

typedef __attribute__((ext_vector_type(16))) __bf16 v16bf;
typedef __attribute__((ext_vector_type(8)))  float  v8f;

// ---------------------------------------------------------------------------
// WMMA helpers (CDNA5 wave32, v_wmma_f32_16x16x32_bf16)
// ---------------------------------------------------------------------------
__device__ __forceinline__ v8f wmma_bf16(v16bf a, v16bf b, v8f c) {
    return __builtin_amdgcn_wmma_f32_16x16x32_bf16(
        /*neg_a=*/false, a, /*neg_b=*/false, b,
        /*c_mod=*/(short)0, c, /*reuse_a=*/false, /*reuse_b=*/false);
}

// A fragment: 16x32 bf16 tile, row-major source (LDS), leading dim ld.
// lane: m = lane&15, hi = lane>>4 ; half j -> k per ISA table.
// (reads two contiguous 16B runs per lane -> compiler emits ds_load_b128 x2)
__device__ __forceinline__ v16bf load_a_frag(const __bf16* p, int ld, int lane, int kBase) {
    const int m  = lane & 15;
    const int hi = lane >> 4;
    v16bf a;
#pragma unroll
    for (int j = 0; j < 16; ++j) {
        int v = j >> 1, pb = j & 1;
        int k = ((v < 4) ? (2 * v) : (16 + 2 * (v - 4))) + pb + 8 * hi;
        a[j] = p[m * ld + kBase + k];
    }
    return a;
}

// B fragment from pre-packed "fragment-native" layout:
// halfs stored as [kt][nt][lane][16]; one contiguous 32B load per lane.
__device__ __forceinline__ v16bf load_b_frag(const __bf16* fragBase, int kt, int nt,
                                             int ntiles, int lane) {
    return *(const v16bf*)(fragBase + (size_t)(((kt * ntiles + nt) * 32 + lane) * 16));
}

// ---------------------------------------------------------------------------
// Weight repack: fp32 [K][N] row-major -> bf16 fragment-native layout
// ---------------------------------------------------------------------------
__global__ __launch_bounds__(256) void k_conv_frag(const float* __restrict__ W,
                                                   __bf16* __restrict__ dst, int K, int N) {
    int idx = blockIdx.x * 256 + threadIdx.x;
    if (idx >= K * N) return;
    int k = idx / N, n = idx % N;
    int kt = k >> 5, kr = k & 31, nt = n >> 4, nr = n & 15;
    int lane = nr + (((kr >> 3) & 1) << 4);
    int j = (kr & 7) + ((kr >> 4) << 3);
    int ntiles = N >> 4;
    dst[(size_t)(((kt * ntiles + nt) * 32 + lane) * 16 + j)] = (__bf16)W[idx];
}

// ---------------------------------------------------------------------------
// x = emb[atom_types-1]   (fp32)
// ---------------------------------------------------------------------------
__global__ __launch_bounds__(256) void k_embed(const int* __restrict__ at,
                                               const float* __restrict__ emb,
                                               float* __restrict__ x) {
    int idx = blockIdx.x * 256 + threadIdx.x;
    if (idx >= N_NODES * HDIM) return;
    int node = idx >> 8, h = idx & 255;
    x[idx] = emb[(at[node] - 1) * HDIM + h];
}

__global__ __launch_bounds__(256) void k_zero(float* __restrict__ p, int n) {
    int idx = blockIdx.x * 256 + threadIdx.x;
    if (idx < n) p[idx] = 0.f;
}

// ---------------------------------------------------------------------------
// xt_h = bf16( x @ nlW + nlb )  : one block = 32 rows x 256 cols, 8 waves.
// Each wave: 2 M-tiles x 2 N-tiles (B frags reused across M).
// ---------------------------------------------------------------------------
__global__ __launch_bounds__(256) void k_node_linear(const float* __restrict__ x,
                                                     const __bf16* __restrict__ Wf,
                                                     const float* __restrict__ bias,
                                                     __bf16* __restrict__ xt_h) {
    __shared__ __bf16 sA[32][HDIM];
    const int t = threadIdx.x;
    const int m0 = blockIdx.x * 32;

    // stage 32 rows of x as bf16
    {
        int r = t >> 3, seg = t & 7;
        const float* src = x + (size_t)(m0 + r) * HDIM + seg * 32;
#pragma unroll
        for (int q = 0; q < 32; ++q) sA[r][seg * 32 + q] = (__bf16)src[q];
    }
    __syncthreads();

    const int w = t >> 5, lane = t & 31;
    const int hi = lane >> 4, nlo = lane & 15;
    const int nt0 = 2 * w, nt1 = 2 * w + 1;
    v8f c[2][2] = {};
#pragma unroll
    for (int kt = 0; kt < HDIM / 32; ++kt) {
        v16bf b0 = load_b_frag(Wf, kt, nt0, 16, lane);
        v16bf b1 = load_b_frag(Wf, kt, nt1, 16, lane);
#pragma unroll
        for (int mt = 0; mt < 2; ++mt) {
            v16bf a = load_a_frag(&sA[mt * 16][0], HDIM, lane, kt * 32);
            c[mt][0] = wmma_bf16(a, b0, c[mt][0]);
            c[mt][1] = wmma_bf16(a, b1, c[mt][1]);
        }
    }
#pragma unroll
    for (int mt = 0; mt < 2; ++mt)
#pragma unroll
        for (int nt = 0; nt < 2; ++nt) {
            int n = (2 * w + nt) * 16 + nlo;
            float b = bias[n];
#pragma unroll
            for (int v = 0; v < 8; ++v)
                xt_h[(size_t)(m0 + mt * 16 + v + 8 * hi) * HDIM + n] = (__bf16)(c[mt][nt][v] + b);
        }
}

// ---------------------------------------------------------------------------
// Fused edge MLP + gate + scatter-add.
// One block = 64 edges (4 M-tiles), 8 waves; each wave: 4 M-tiles x 2 N-tiles.
// GEMM1 (K=512) -> relu -> GEMM2 (K=256) -> *ea -> atomicAdd into agg.
// ---------------------------------------------------------------------------
__global__ __launch_bounds__(256) void k_edge(const __bf16* __restrict__ xt_h,
                                              const int* __restrict__ ei,
                                              const float* __restrict__ dist,
                                              const __bf16* __restrict__ m1f,
                                              const float* __restrict__ m1b,
                                              const __bf16* __restrict__ m2f,
                                              const float* __restrict__ m2b,
                                              const float* __restrict__ elW,
                                              const float* __restrict__ elb,
                                              float* __restrict__ agg) {
    __shared__ __bf16 sA[64][2 * HDIM];   // [dst feat | src feat]  (64 KB)
    __shared__ __bf16 sH[64][HDIM];       // hidden, post-relu bf16 (32 KB)
    __shared__ int    sDst[64];
    __shared__ float  sDist[64];

    const int t = threadIdx.x;
    const int e0 = blockIdx.x * 64;

    if (t < 64) { sDst[t] = ei[N_EDGES + e0 + t]; sDist[t] = dist[e0 + t]; }

    // gather: each thread copies 128 halfs (256B) of a concat row
    {
        int r = t >> 2, seg = t & 3;
        int e = e0 + r;
        int node = (seg < 2) ? ei[N_EDGES + e] : ei[e];   // [xt[dst], xt[src]]
        const uint4* sp = (const uint4*)(xt_h + (size_t)node * HDIM + (seg & 1) * 128);
        uint4* dp = (uint4*)(&sA[r][seg * 128]);
#pragma unroll
        for (int q = 0; q < 16; ++q) dp[q] = sp[q];
    }
    __syncthreads();

    const int w = t >> 5, lane = t & 31;
    const int hi = lane >> 4, nlo = lane & 15;

    // ---- GEMM1: hidden = relu(cat @ m1W + m1b), K = 512
    {
        v8f c[4][2] = {};
#pragma unroll
        for (int kt = 0; kt < (2 * HDIM) / 32; ++kt) {
            v16bf b0 = load_b_frag(m1f, kt, 2 * w, 16, lane);
            v16bf b1 = load_b_frag(m1f, kt, 2 * w + 1, 16, lane);
#pragma unroll
            for (int mt = 0; mt < 4; ++mt) {
                v16bf a = load_a_frag(&sA[mt * 16][0], 2 * HDIM, lane, kt * 32);
                c[mt][0] = wmma_bf16(a, b0, c[mt][0]);
                c[mt][1] = wmma_bf16(a, b1, c[mt][1]);
            }
        }
#pragma unroll
        for (int mt = 0; mt < 4; ++mt)
#pragma unroll
            for (int nt = 0; nt < 2; ++nt) {
                int n = (2 * w + nt) * 16 + nlo;
                float b = m1b[n];
#pragma unroll
                for (int v = 0; v < 8; ++v)
                    sH[mt * 16 + v + 8 * hi][n] = (__bf16)fmaxf(c[mt][nt][v] + b, 0.f);
            }
    }
    __syncthreads();

    // ---- GEMM2: msg = hidden @ m2W + m2b, K = 256 ; gate; scatter-add
    {
        v8f d[4][2] = {};
#pragma unroll
        for (int kt = 0; kt < HDIM / 32; ++kt) {
            v16bf b0 = load_b_frag(m2f, kt, 2 * w, 16, lane);
            v16bf b1 = load_b_frag(m2f, kt, 2 * w + 1, 16, lane);
#pragma unroll
            for (int mt = 0; mt < 4; ++mt) {
                v16bf a = load_a_frag(&sH[mt * 16][0], HDIM, lane, kt * 32);
                d[mt][0] = wmma_bf16(a, b0, d[mt][0]);
                d[mt][1] = wmma_bf16(a, b1, d[mt][1]);
            }
        }
#pragma unroll
        for (int nt = 0; nt < 2; ++nt) {
            int n = (2 * w + nt) * 16 + nlo;
            float b = m2b[n], we = elW[n], be = elb[n];
#pragma unroll
            for (int mt = 0; mt < 4; ++mt)
#pragma unroll
                for (int v = 0; v < 8; ++v) {
                    int m = mt * 16 + v + 8 * hi;
                    float val = (d[mt][nt][v] + b) * (sDist[m] * we + be);
                    atomicAdd(&agg[(size_t)sDst[m] * HDIM + n], val);
                }
        }
    }
}

// ---------------------------------------------------------------------------
// Fused node update: h = relu([agg,x]@u1W+u1b); xn = relu(h@u2W+u2b);
// xn = LN(xn)*g+b ; x = (layer>0 ? x+xn : xn).  One block = 32 nodes, 8 waves,
// each wave: 2 M-tiles x 2 N-tiles.
// ---------------------------------------------------------------------------
__global__ __launch_bounds__(256) void k_node_update(const float* __restrict__ agg,
                                                     float* __restrict__ x,
                                                     const __bf16* __restrict__ u1f,
                                                     const float* __restrict__ u1b,
                                                     const __bf16* __restrict__ u2f,
                                                     const float* __restrict__ u2b,
                                                     const float* __restrict__ lnG,
                                                     const float* __restrict__ lnB,
                                                     int layer) {
    __shared__ __bf16 sA[32][2 * HDIM];   // 32 KB
    __shared__ __bf16 sH[32][HDIM];       // 16 KB
    __shared__ float  sO[32][HDIM];       // 32 KB
    __shared__ float  sMu[32], sRs[32];

    const int t = threadIdx.x;
    const int n0 = blockIdx.x * 32;

    // stage cat([agg, x]) as bf16 : 32x512 halfs, 64 per thread
    {
        int r = t >> 3, seg = t & 7;
        size_t row = (size_t)(n0 + r) * HDIM;
#pragma unroll
        for (int q = 0; q < 64; ++q) {
            int col = seg * 64 + q;
            float v = (col < HDIM) ? agg[row + col] : x[row + col - HDIM];
            sA[r][col] = (__bf16)v;
        }
    }
    __syncthreads();

    const int w = t >> 5, lane = t & 31;
    const int hi = lane >> 4, nlo = lane & 15;

    {
        v8f c[2][2] = {};
#pragma unroll
        for (int kt = 0; kt < (2 * HDIM) / 32; ++kt) {
            v16bf b0 = load_b_frag(u1f, kt, 2 * w, 16, lane);
            v16bf b1 = load_b_frag(u1f, kt, 2 * w + 1, 16, lane);
#pragma unroll
            for (int mt = 0; mt < 2; ++mt) {
                v16bf a = load_a_frag(&sA[mt * 16][0], 2 * HDIM, lane, kt * 32);
                c[mt][0] = wmma_bf16(a, b0, c[mt][0]);
                c[mt][1] = wmma_bf16(a, b1, c[mt][1]);
            }
        }
#pragma unroll
        for (int mt = 0; mt < 2; ++mt)
#pragma unroll
            for (int nt = 0; nt < 2; ++nt) {
                int n = (2 * w + nt) * 16 + nlo;
                float b = u1b[n];
#pragma unroll
                for (int v = 0; v < 8; ++v)
                    sH[mt * 16 + v + 8 * hi][n] = (__bf16)fmaxf(c[mt][nt][v] + b, 0.f);
            }
    }
    __syncthreads();

    {
        v8f d[2][2] = {};
#pragma unroll
        for (int kt = 0; kt < HDIM / 32; ++kt) {
            v16bf b0 = load_b_frag(u2f, kt, 2 * w, 16, lane);
            v16bf b1 = load_b_frag(u2f, kt, 2 * w + 1, 16, lane);
#pragma unroll
            for (int mt = 0; mt < 2; ++mt) {
                v16bf a = load_a_frag(&sH[mt * 16][0], HDIM, lane, kt * 32);
                d[mt][0] = wmma_bf16(a, b0, d[mt][0]);
                d[mt][1] = wmma_bf16(a, b1, d[mt][1]);
            }
        }
#pragma unroll
        for (int mt = 0; mt < 2; ++mt)
#pragma unroll
            for (int nt = 0; nt < 2; ++nt) {
                int n = (2 * w + nt) * 16 + nlo;
                float b = u2b[n];
#pragma unroll
                for (int v = 0; v < 8; ++v)
                    sO[mt * 16 + v + 8 * hi][n] = fmaxf(d[mt][nt][v] + b, 0.f);
            }
    }
    __syncthreads();

    // per-row layernorm stats
    if (t < 32) {
        float s = 0.f, s2 = 0.f;
        for (int n = 0; n < HDIM; ++n) { float v = sO[t][n]; s += v; s2 += v * v; }
        float mu = s * (1.f / HDIM);
        float var = s2 * (1.f / HDIM) - mu * mu;
        sMu[t] = mu;
        sRs[t] = rsqrtf(var + 1e-5f);
    }
    __syncthreads();

    {
        int r = t >> 3, cbase = (t & 7) * 32;
        size_t row = (size_t)(n0 + r) * HDIM;
        float mu = sMu[r], rs = sRs[r];
#pragma unroll
        for (int q = 0; q < 32; ++q) {
            int n = cbase + q;
            float v = (sO[r][n] - mu) * rs * lnG[n] + lnB[n];
            float o = (layer > 0) ? (x[row + n] + v) : v;
            x[row + n] = o;
        }
    }
}

// ---------------------------------------------------------------------------
// Graph pooling (segment mean) + MLP head
// ---------------------------------------------------------------------------
__global__ __launch_bounds__(256) void k_pool(const float* __restrict__ x,
                                              const int* __restrict__ bid,
                                              float* __restrict__ sums,
                                              float* __restrict__ counts) {
    int idx = blockIdx.x * 256 + threadIdx.x;
    if (idx >= N_NODES * HDIM) return;
    int node = idx >> 8, h = idx & 255;
    int b = bid[node];
    atomicAdd(&sums[b * HDIM + h], x[idx]);
    if (h == 0) atomicAdd(&counts[b], 1.f);
}

__global__ __launch_bounds__(256) void k_head(const float* __restrict__ sums,
                                              const float* __restrict__ counts,
                                              const float* __restrict__ pW1, const float* __restrict__ pb1,
                                              const float* __restrict__ pW2, const float* __restrict__ pb2,
                                              const float* __restrict__ pW3, const float* __restrict__ pb3,
                                              const float* __restrict__ pW4, const float* __restrict__ pb4,
                                              float* __restrict__ out) {
    __shared__ float sP[N_GRAPHS][256];
    __shared__ float s1[N_GRAPHS][256];
    __shared__ float s2[N_GRAPHS][128];
    __shared__ float s3[N_GRAPHS][64];
    const int t = threadIdx.x;
    for (int i = t; i < N_GRAPHS * 256; i += 256) {
        int g = i >> 8, h = i & 255;
        float c = counts[g]; c = (c < 1.f) ? 1.f : c;
        sP[g][h] = sums[i] / c;
    }
    __syncthreads();
    for (int i = t; i < N_GRAPHS * 256; i += 256) {
        int g = i >> 8, h = i & 255;
        float acc = pb1[h];
        for (int k = 0; k < 256; ++k) acc += sP[g][k] * pW1[k * 256 + h];
        s1[g][h] = fmaxf(acc, 0.f);
    }
    __syncthreads();
    for (int i = t; i < N_GRAPHS * 128; i += 256) {
        int g = i >> 7, h = i & 127;
        float acc = pb2[h];
        for (int k = 0; k < 256; ++k) acc += s1[g][k] * pW2[k * 128 + h];
        s2[g][h] = fmaxf(acc, 0.f);
    }
    __syncthreads();
    for (int i = t; i < N_GRAPHS * 64; i += 256) {
        int g = i >> 6, h = i & 63;
        float acc = pb3[h];
        for (int k = 0; k < 128; ++k) acc += s2[g][k] * pW3[k * 64 + h];
        s3[g][h] = fmaxf(acc, 0.f);
    }
    __syncthreads();
    if (t < N_GRAPHS) {
        float acc = pb4[0];
        for (int k = 0; k < 64; ++k) acc += s3[t][k] * pW4[k];
        out[t] = acc;
    }
}

// ---------------------------------------------------------------------------
// Host launcher
// ---------------------------------------------------------------------------
extern "C" void kernel_launch(void* const* d_in, const int* in_sizes, int n_in,
                              void* d_out, int out_size, void* d_ws, size_t ws_size,
                              hipStream_t stream) {
    const int*   atom_types = (const int*)d_in[0];
    const int*   ei   = (const int*)d_in[1];
    const int*   bid  = (const int*)d_in[2];
    const float* dist = (const float*)d_in[3];
    const float* emb  = (const float*)d_in[4];
    const float* nlW  = (const float*)d_in[5];
    const float* nlb  = (const float*)d_in[6];
    const float* elW  = (const float*)d_in[7];
    const float* elb  = (const float*)d_in[8];
    const float* m1W  = (const float*)d_in[9];
    const float* m1b  = (const float*)d_in[10];
    const float* m2W  = (const float*)d_in[11];
    const float* m2b  = (const float*)d_in[12];
    const float* u1W  = (const float*)d_in[13];
    const float* u1b  = (const float*)d_in[14];
    const float* u2W  = (const float*)d_in[15];
    const float* u2b  = (const float*)d_in[16];
    const float* lnG  = (const float*)d_in[17];
    const float* lnB  = (const float*)d_in[18];
    const float* pW1  = (const float*)d_in[19];
    const float* pb1  = (const float*)d_in[20];
    const float* pW2  = (const float*)d_in[21];
    const float* pb2  = (const float*)d_in[22];
    const float* pW3  = (const float*)d_in[23];
    const float* pb3  = (const float*)d_in[24];
    const float* pW4  = (const float*)d_in[25];
    const float* pb4  = (const float*)d_in[26];
    float* out = (float*)d_out;

    // workspace carve-out (256B aligned)
    char* ws = (char*)d_ws;
    size_t off = 0;
    auto carve = [&](size_t bytes) -> char* {
        char* p = ws + off;
        off += (bytes + 255) & ~(size_t)255;
        return p;
    };
    float*  x     = (float*)carve((size_t)N_NODES * HDIM * 4);
    __bf16* xt_h  = (__bf16*)carve((size_t)N_NODES * HDIM * 2);
    float*  agg   = (float*)carve((size_t)N_NODES * HDIM * 4);
    const size_t LAYER_FRAG = (size_t)(HDIM + 2 * HDIM + HDIM + 2 * HDIM + HDIM) * HDIM; // halfs
    __bf16* frag  = (__bf16*)carve(LAYER_FRAG * NLAYERS * 2);
    float*  sums  = (float*)carve((size_t)N_GRAPHS * HDIM * 4);
    float*  cnts  = (float*)carve((size_t)N_GRAPHS * 4);
    (void)ws_size; (void)in_sizes; (void)n_in; (void)out_size;

    // ---- repack weights to bf16 fragment-native layout (once per call)
    for (int i = 0; i < NLAYERS; ++i) {
        __bf16* nl_f = frag + (size_t)i * LAYER_FRAG;
        __bf16* m1_f = nl_f + (size_t)HDIM * HDIM;
        __bf16* m2_f = m1_f + (size_t)2 * HDIM * HDIM;
        __bf16* u1_f = m2_f + (size_t)HDIM * HDIM;
        __bf16* u2_f = u1_f + (size_t)2 * HDIM * HDIM;
        k_conv_frag<<<(HDIM * HDIM + 255) / 256, 256, 0, stream>>>(nlW + (size_t)i * HDIM * HDIM, nl_f, HDIM, HDIM);
        k_conv_frag<<<(2 * HDIM * HDIM + 255) / 256, 256, 0, stream>>>(m1W + (size_t)i * 2 * HDIM * HDIM, m1_f, 2 * HDIM, HDIM);
        k_conv_frag<<<(HDIM * HDIM + 255) / 256, 256, 0, stream>>>(m2W + (size_t)i * HDIM * HDIM, m2_f, HDIM, HDIM);
        k_conv_frag<<<(2 * HDIM * HDIM + 255) / 256, 256, 0, stream>>>(u1W + (size_t)i * 2 * HDIM * HDIM, u1_f, 2 * HDIM, HDIM);
        k_conv_frag<<<(HDIM * HDIM + 255) / 256, 256, 0, stream>>>(u2W + (size_t)i * HDIM * HDIM, u2_f, HDIM, HDIM);
    }

    k_zero<<<(N_GRAPHS * HDIM + 255) / 256, 256, 0, stream>>>(sums, N_GRAPHS * HDIM);
    k_zero<<<1, 256, 0, stream>>>(cnts, N_GRAPHS);
    k_embed<<<(N_NODES * HDIM + 255) / 256, 256, 0, stream>>>(atom_types, emb, x);

    for (int i = 0; i < NLAYERS; ++i) {
        __bf16* nl_f = frag + (size_t)i * LAYER_FRAG;
        __bf16* m1_f = nl_f + (size_t)HDIM * HDIM;
        __bf16* m2_f = m1_f + (size_t)2 * HDIM * HDIM;
        __bf16* u1_f = m2_f + (size_t)HDIM * HDIM;
        __bf16* u2_f = u1_f + (size_t)2 * HDIM * HDIM;

        k_node_linear<<<N_NODES / 32, 256, 0, stream>>>(x, nl_f, nlb + i * HDIM, xt_h);
        k_zero<<<(N_NODES * HDIM + 255) / 256, 256, 0, stream>>>(agg, N_NODES * HDIM);
        k_edge<<<N_EDGES / 64, 256, 0, stream>>>(xt_h, ei, dist,
                                                 m1_f, m1b + i * HDIM,
                                                 m2_f, m2b + i * HDIM,
                                                 elW + i * HDIM, elb + i * HDIM, agg);
        k_node_update<<<N_NODES / 32, 256, 0, stream>>>(agg, x,
                                                        u1_f, u1b + i * HDIM,
                                                        u2_f, u2b + i * HDIM,
                                                        lnG + i * HDIM, lnB + i * HDIM, i);
    }

    k_pool<<<(N_NODES * HDIM + 255) / 256, 256, 0, stream>>>(x, bid, sums, cnts);
    k_head<<<1, 256, 0, stream>>>(sums, cnts, pW1, pb1, pW2, pb2, pW3, pb3, pW4, pb4, out);
}
